// GATv2_52450140619486
// MI455X (gfx1250) — matrix operand order repaired and checked by reading, not compiled, not run on hardware
//
#include <hip/hip_runtime.h>
#include <math.h>

#define N_NODES 10000
#define N_EDGES 160000
#define EPRIME  (N_EDGES + N_NODES)   // edges + self loops
#define NHEAD   7
#define CDIM    64
#define HC      (NHEAD * CDIM)        // 448

typedef __bf16 bf16_t;
typedef bf16_t  v16bf __attribute__((ext_vector_type(16)));
typedef float   v8f   __attribute__((ext_vector_type(8)));
typedef unsigned u32x4 __attribute__((ext_vector_type(4)));

union FragU { v16bf v; u32x4 q[2]; unsigned short u[16]; };

__device__ __forceinline__ unsigned short f2bf(float f) {
  unsigned u = __float_as_uint(f);
  u += 0x7fffu + ((u >> 16) & 1u);          // round-to-nearest-even
  return (unsigned short)(u >> 16);
}
__device__ __forceinline__ float bf2f(unsigned short h) {
  return __uint_as_float(((unsigned)h) << 16);
}

// ---------------------------------------------------------------- utilities
__global__ void cvt_f32_to_bf16(const float* __restrict__ in,
                                unsigned short* __restrict__ out, int n) {
  int i = blockIdx.x * blockDim.x + threadIdx.x;
  if (i < n) out[i] = f2bf(in[i]);
}

__global__ void fill_f32(float* __restrict__ p, float v, int n) {
  int i = blockIdx.x * blockDim.x + threadIdx.x;
  if (i < n) p[i] = v;
}

// Repack weight W[K,N] (f32, row-major) into bf16 WMMA-B fragment order:
// tile (kt,nt) of 32x16 stored as lane-major: 32 lanes x 16 contiguous bf16,
// where lane's element e is  k = kt*32 + (lane>>4)*8 + (e&7) + (e>>3)*16, col = nt*16+(lane&15)
// (CDNA5 ISA 7.12.2, 16-bit B-matrix 32x16). GEMM B loads become 2x b128/lane.
__global__ void repack_w_bf16(const float* __restrict__ W,
                              unsigned short* __restrict__ Bp, int K, int N) {
  int i = blockIdx.x * blockDim.x + threadIdx.x;
  if (i >= K * N) return;
  int e    = i & 15;
  int lane = (i >> 4) & 31;
  int tile = i >> 9;
  int ntiles = N >> 4;
  int nt = tile % ntiles, kt = tile / ntiles;
  int k   = kt * 32 + (lane >> 4) * 8 + (e & 7) + (e >> 3) * 16;
  int col = nt * 16 + (lane & 15);
  Bp[i] = f2bf(W[(size_t)k * N + col]);
}

// ------------------------------------------------- bf16 WMMA GEMM (fp32 acc)
// C[M,N] = act(A[M,K] @ B[K,N] + bias). One wave -> 16 x (16*NT) strip.
// A: bf16 row-major (K mult of 32). Bp: fragment-packed (repack_w_bf16).
template <int NT>
__global__ void wmma_gemm_bf16(const unsigned short* __restrict__ A,
                               const unsigned short* __restrict__ Bp,
                               const float* __restrict__ bias,
                               float* __restrict__ outF,
                               unsigned short* __restrict__ outB,
                               int M, int N, int K, int act) {
  const int lane = threadIdx.x & 31;
  const int half = lane >> 4;   // 0: lanes 0-15, 1: lanes 16-31
  const int l15  = lane & 15;
  const int tm   = blockIdx.y;
  const int tn0  = blockIdx.x * NT;
  const int ntiles = N >> 4;

  const int arow = tm * 16 + l15;
  const unsigned short* __restrict__ Arow = A + (size_t)arow * K;

  v8f acc[NT];
  const v8f vzero = {0.f, 0.f, 0.f, 0.f, 0.f, 0.f, 0.f, 0.f};
  #pragma unroll
  for (int nt = 0; nt < NT; ++nt) acc[nt] = vzero;

  const int nk = K >> 5;
  for (int kt = 0; kt < nk; ++kt) {
    // A fragment: lane holds rows of two contiguous 8-elem runs (16B aligned)
    FragU a;
    const u32x4* __restrict__ pa =
        (const u32x4*)(Arow + (kt << 5) + half * 8);
    a.q[0] = pa[0];   // K = k0+half*8 .. +7
    a.q[1] = pa[2];   // K = k0+16+half*8 .. +7   (+32 bytes)
    if (kt + 1 < nk) __builtin_prefetch(Arow + ((kt + 1) << 5), 0, 1);

    #pragma unroll
    for (int nt = 0; nt < NT; ++nt) {
      FragU b;
      const u32x4* __restrict__ pb = (const u32x4*)(
          Bp + (((size_t)(kt * ntiles + tn0 + nt) * 32 + lane) << 4));
      b.q[0] = pb[0];
      b.q[1] = pb[1];
      acc[nt] = __builtin_amdgcn_wmma_f32_16x16x32_bf16(
          /*neg_a=*/false, a.v, /*neg_b=*/false, b.v,
          /*c_mod=*/(short)0, acc[nt], /*reuse_a=*/false, /*reuse_b=*/false);
    }
  }

  // C/D: VGPR v holds row (v + 8*half), column lane&15.
  #pragma unroll
  for (int nt = 0; nt < NT; ++nt) {
    const int bcol = (tn0 + nt) * 16 + l15;
    const float bv = bias ? bias[bcol] : 0.f;
    #pragma unroll
    for (int v = 0; v < 8; ++v) {
      int row = tm * 16 + v + half * 8;
      float val = acc[nt][v] + bv;
      if (act) val = fmaxf(val, 0.f);
      size_t idx = (size_t)row * N + bcol;
      if (outF) outF[idx] = val;
      if (outB) outB[idx] = f2bf(val);
    }
  }
}

// ---------------------------------------------------------------- GAT edges
__device__ __forceinline__ void atomicMaxF(float* addr, float val) {
  unsigned* ua = (unsigned*)addr;
  unsigned old = *ua;
  while (__uint_as_float(old) < val) {
    unsigned assumed = old;
    old = atomicCAS(ua, assumed, __float_as_uint(val));
    if (old == assumed) break;
  }
}

__device__ __forceinline__ void edge_ids(const int* __restrict__ ei, int e,
                                         int& src, int& dst) {
  if (e < N_EDGES) { src = ei[e]; dst = ei[N_EDGES + e]; }
  else             { src = e - N_EDGES; dst = e - N_EDGES; }
}

// logits[e,h] = sum_c leaky(xl[src,h,c] + xr[dst,h,c]) * att[h,c]; seg-max -> m
__global__ void edge_logits(const int* __restrict__ ei,
                            const float* __restrict__ xl,
                            const float* __restrict__ xr,
                            const float* __restrict__ att,   // [H,C] layer slice
                            float* __restrict__ logits,      // [E',H]
                            float* __restrict__ m) {         // [N,H] init -inf
  int idx = blockIdx.x * blockDim.x + threadIdx.x;
  if (idx >= EPRIME * NHEAD) return;
  int e = idx / NHEAD, h = idx - e * NHEAD;
  int src, dst; edge_ids(ei, e, src, dst);
  const float* pl = xl + (size_t)src * HC + h * CDIM;
  const float* pr = xr + (size_t)dst * HC + h * CDIM;
  const float* pa = att + h * CDIM;
  float s = 0.f;
  #pragma unroll 8
  for (int c = 0; c < CDIM; ++c) {
    float t = pl[c] + pr[c];
    t = (t > 0.f) ? t : 0.2f * t;             // leaky_relu(0.2)
    s += t * pa[c];
  }
  logits[idx] = s;
  atomicMaxF(&m[dst * NHEAD + h], s);
}

// a = exp(logit - m[dst]); seg-sum -> ssum; overwrite logits with a
__global__ void edge_expsum(const int* __restrict__ ei,
                            float* __restrict__ logits,
                            const float* __restrict__ m,
                            float* __restrict__ ssum) {      // [N,H] init 0
  int idx = blockIdx.x * blockDim.x + threadIdx.x;
  if (idx >= EPRIME * NHEAD) return;
  int e = idx / NHEAD, h = idx - e * NHEAD;
  int src, dst; edge_ids(ei, e, src, dst);
  (void)src;
  float a = __expf(logits[idx] - m[dst * NHEAD + h]);
  logits[idx] = a;
  atomicAdd(&ssum[dst * NHEAD + h], a);
}

// accum[dst,h,:] += (a / ssum[dst,h]) * xl[src,h,:]
__global__ void edge_scatter(const int* __restrict__ ei,
                             const float* __restrict__ logits,
                             const float* __restrict__ ssum,
                             const float* __restrict__ xl,
                             float* __restrict__ accum) {    // [N,H,C] init 0
  int idx = blockIdx.x * blockDim.x + threadIdx.x;
  if (idx >= EPRIME * NHEAD) return;
  int e = idx / NHEAD, h = idx - e * NHEAD;
  int src, dst; edge_ids(ei, e, src, dst);
  float alpha = logits[idx] / (ssum[dst * NHEAD + h] + 1e-16f);
  const float* pl = xl + (size_t)src * HC + h * CDIM;
  float* pd = accum + (size_t)dst * HC + h * CDIM;
  #pragma unroll 4
  for (int c = 0; c < CDIM; ++c) atomicAdd(&pd[c], alpha * pl[c]);
}

// h_new = relu(mean_h(accum) + cbias) -> bf16 for the next WMMA GEMM
__global__ void gat_finalize(const float* __restrict__ accum,
                             const float* __restrict__ cbias,  // [C]
                             unsigned short* __restrict__ hb) {
  int i = blockIdx.x * blockDim.x + threadIdx.x;  // n*64 + c
  if (i >= N_NODES * CDIM) return;
  int n = i >> 6, c = i & 63;
  float s = 0.f;
  #pragma unroll
  for (int h = 0; h < NHEAD; ++h) s += accum[(size_t)n * HC + h * CDIM + c];
  float v = s * (1.f / 7.f) + cbias[c];
  hb[i] = f2bf(fmaxf(v, 0.f));
}

// final 16->9 layer + log_softmax (too small/ragged for WMMA)
__global__ void head_logsoftmax(const unsigned short* __restrict__ g2,  // [N,16]
                                const float* __restrict__ V3,           // [16,9]
                                const float* __restrict__ c3,           // [9]
                                float* __restrict__ out) {              // [N,9]
  int n = blockIdx.x * blockDim.x + threadIdx.x;
  if (n >= N_NODES) return;
  float in[16];
  #pragma unroll
  for (int k = 0; k < 16; ++k) in[k] = bf2f(g2[n * 16 + k]);
  float z[9]; float mx = -1e30f;
  #pragma unroll
  for (int j = 0; j < 9; ++j) {
    float s = c3[j];
    #pragma unroll
    for (int k = 0; k < 16; ++k) s += in[k] * V3[k * 9 + j];
    z[j] = s; mx = fmaxf(mx, s);
  }
  float se = 0.f;
  #pragma unroll
  for (int j = 0; j < 9; ++j) se += __expf(z[j] - mx);
  float lse = mx + __logf(se);
  #pragma unroll
  for (int j = 0; j < 9; ++j) out[n * 9 + j] = z[j] - lse;
}

// ------------------------------------------------------------------- driver
extern "C" void kernel_launch(void* const* d_in, const int* in_sizes, int n_in,
                              void* d_out, int out_size, void* d_ws, size_t ws_size,
                              hipStream_t stream) {
  const float* x   = (const float*)d_in[0];
  const int*   ei  = (const int*)  d_in[1];
  const float* W1  = (const float*)d_in[2];  const float* b1 = (const float*)d_in[3];
  const float* W2  = (const float*)d_in[4];  const float* b2 = (const float*)d_in[5];
  const float* W3  = (const float*)d_in[6];  const float* b3 = (const float*)d_in[7];
  const float* Wl  = (const float*)d_in[8];  const float* bl = (const float*)d_in[9];
  const float* Wr  = (const float*)d_in[10]; const float* att= (const float*)d_in[11];
  const float* cb  = (const float*)d_in[12];
  const float* V1  = (const float*)d_in[13]; const float* c1 = (const float*)d_in[14];
  const float* V2  = (const float*)d_in[15]; const float* c2 = (const float*)d_in[16];
  const float* V3  = (const float*)d_in[17]; const float* c3 = (const float*)d_in[18];
  float* out = (float*)d_out;
  (void)in_sizes; (void)n_in; (void)out_size; (void)ws_size;

  char* ws = (char*)d_ws; size_t off = 0;
  auto alloc = [&](size_t bytes) -> void* {
    void* p = ws + off;
    off += (bytes + 255) & ~(size_t)255;
    return p;
  };
  // bf16 staging buffers (weights in fragment-packed order)
  unsigned short* xb  = (unsigned short*)alloc((size_t)N_NODES * 1024 * 2);
  unsigned short* W1p = (unsigned short*)alloc((size_t)1024 * 512 * 2);
  unsigned short* W2p = (unsigned short*)alloc((size_t)512 * 256 * 2);
  unsigned short* W3p = (unsigned short*)alloc((size_t)256 * 64 * 2);
  unsigned short* Wlp = (unsigned short*)alloc((size_t)5 * 64 * HC * 2);
  unsigned short* Wrp = (unsigned short*)alloc((size_t)5 * 64 * HC * 2);
  unsigned short* V1p = (unsigned short*)alloc((size_t)64 * 32 * 2);
  unsigned short* V2p = (unsigned short*)alloc((size_t)32 * 16 * 2);
  unsigned short* h1b = (unsigned short*)alloc((size_t)N_NODES * 512 * 2);
  unsigned short* h2b = (unsigned short*)alloc((size_t)N_NODES * 256 * 2);
  unsigned short* hb  = (unsigned short*)alloc((size_t)N_NODES * 64 * 2);
  unsigned short* g1b = (unsigned short*)alloc((size_t)N_NODES * 32 * 2);
  unsigned short* g2b = (unsigned short*)alloc((size_t)N_NODES * 16 * 2);
  // fp32 GAT scratch
  float* xl    = (float*)alloc((size_t)N_NODES * HC * 4);
  float* xr    = (float*)alloc((size_t)N_NODES * HC * 4);
  float* lgt   = (float*)alloc((size_t)EPRIME * NHEAD * 4);
  float* mbuf  = (float*)alloc((size_t)N_NODES * NHEAD * 4);
  float* sbuf  = (float*)alloc((size_t)N_NODES * NHEAD * 4);
  float* accum = (float*)alloc((size_t)N_NODES * HC * 4);

  auto cvt = [&](const float* in, unsigned short* o, int n) {
    cvt_f32_to_bf16<<<dim3((n + 255) / 256), dim3(256), 0, stream>>>(in, o, n);
  };
  auto repack = [&](const float* W, unsigned short* Bp, int K, int Nn) {
    int n = K * Nn;
    repack_w_bf16<<<dim3((n + 255) / 256), dim3(256), 0, stream>>>(W, Bp, K, Nn);
  };
  auto fill = [&](float* p, float v, int n) {
    fill_f32<<<dim3((n + 255) / 256), dim3(256), 0, stream>>>(p, v, n);
  };
  auto gemm = [&](const unsigned short* A, const unsigned short* Bp,
                  const float* bias, float* oF, unsigned short* oB,
                  int M, int Nn, int K, int act) {
    if ((Nn & 63) == 0)
      wmma_gemm_bf16<4><<<dim3(Nn / 64, M / 16), dim3(32), 0, stream>>>(
          A, Bp, bias, oF, oB, M, Nn, K, act);
    else
      wmma_gemm_bf16<1><<<dim3(Nn / 16, M / 16), dim3(32), 0, stream>>>(
          A, Bp, bias, oF, oB, M, Nn, K, act);
  };

  // precision staging + weight repack (recomputed every call; deterministic)
  cvt(x, xb, N_NODES * 1024);
  repack(W1, W1p, 1024, 512);
  repack(W2, W2p, 512, 256);
  repack(W3, W3p, 256, 64);
  for (int l = 0; l < 5; ++l) {
    repack(Wl + (size_t)l * 64 * HC, Wlp + (size_t)l * 64 * HC, 64, HC);
    repack(Wr + (size_t)l * 64 * HC, Wrp + (size_t)l * 64 * HC, 64, HC);
  }
  repack(V1, V1p, 64, 32);
  repack(V2, V2p, 32, 16);

  // input MLP: 1024 -> 512 -> 256 -> 64, ReLU each (WMMA bf16, fp32 acc)
  gemm(xb,  W1p, b1, nullptr, h1b, N_NODES, 512, 1024, 1);
  gemm(h1b, W2p, b2, nullptr, h2b, N_NODES, 256, 512,  1);
  gemm(h2b, W3p, b3, nullptr, hb,  N_NODES, 64,  256,  1);

  const int EH = EPRIME * NHEAD;
  for (int l = 0; l < 5; ++l) {
    // xl = h@Wl + bl ; xr = h@Wr   (fp32 outputs for the edge math)
    gemm(hb, Wlp + (size_t)l * 64 * HC, bl + (size_t)l * HC, xl, nullptr,
         N_NODES, HC, 64, 0);
    gemm(hb, Wrp + (size_t)l * 64 * HC, nullptr, xr, nullptr,
         N_NODES, HC, 64, 0);
    fill(mbuf,  -1e30f, N_NODES * NHEAD);
    fill(sbuf,  0.f,    N_NODES * NHEAD);
    fill(accum, 0.f,    N_NODES * HC);
    edge_logits<<<dim3((EH + 255) / 256), dim3(256), 0, stream>>>(
        ei, xl, xr, att + (size_t)l * HC, lgt, mbuf);
    edge_expsum<<<dim3((EH + 255) / 256), dim3(256), 0, stream>>>(
        ei, lgt, mbuf, sbuf);
    edge_scatter<<<dim3((EH + 255) / 256), dim3(256), 0, stream>>>(
        ei, lgt, sbuf, xl, accum);
    gat_finalize<<<dim3((N_NODES * CDIM + 255) / 256), dim3(256), 0, stream>>>(
        accum, cb + (size_t)l * CDIM, hb);
  }

  // output MLP: 64 -> 32 -> 16 (WMMA), then 16 -> 9 + log_softmax (scalar)
  gemm(hb,  V1p, c1, nullptr, g1b, N_NODES, 32, 64, 1);
  gemm(g1b, V2p, c2, nullptr, g2b, N_NODES, 16, 32, 1);
  head_logsoftmax<<<dim3((N_NODES + 255) / 256), dim3(256), 0, stream>>>(
      g2b, V3, c3, out);
}
